// Seq_Encode_39187281609310
// MI455X (gfx1250) — matrix operand order repaired and checked by reading, not compiled, not run on hardware
//
#include <hip/hip_runtime.h>
#include <cstddef>
#include <cstdint>

// ---------------- problem constants (from the reference) ----------------
constexpr int E     = 768;
constexpr int Hh    = 2;
constexpr int HDim  = 384;     // E / H
constexpr int FFdim = 256;
constexpr int Lnum  = 3;
constexpr int ENC   = 128;
constexpr int INDIM = 234;
constexpr int Bb    = 16;
constexpr int Ss    = 512;
constexpr int Mrows = Bb * Ss; // 8192
constexpr float NEG = -1e9f;

// ---------------- WMMA / TDM types ----------------
typedef __attribute__((ext_vector_type(16))) __bf16        v16bf;
typedef __attribute__((ext_vector_type(8)))  float         v8f;
typedef __attribute__((ext_vector_type(4)))  unsigned int  u32x4;
typedef __attribute__((ext_vector_type(8)))  unsigned int  u32x8;

__device__ __forceinline__ __bf16 to_bf16(float f) { return (__bf16)f; }

__device__ __forceinline__ v8f wmma_bf16(v16bf a, v16bf b, v8f c) {
    // D = A(16x32 bf16) x B(32x16 bf16) + C(16x16 f32)
    return __builtin_amdgcn_wmma_f32_16x16x32_bf16(
        /*neg_a=*/false, a, /*neg_b=*/false, b,
        /*c_mod=*/(short)0, c, /*reuse_a=*/false, /*reuse_b=*/false);
}

// =======================================================================
// Tensor Data Mover: 2D tile load (f32 elements) global -> LDS.
// D# per CDNA5 ISA:
//   group0: count=1 | lds_addr | global_addr | type=2
//   group1: data_size=4B, pad_enable (pad 2 DWORDs per 32 DWORDs -> LDS row
//           stride 34 dwords, bank-conflict free), tensor_dim0/1 for OOB
//           zero-fill of the K tail, tile_dim0=32 (K slab), tile_dim1=rows,
//           tensor_dim0_stride = row stride in elements.
// Wave-level op (EXEC ignored); issue from one wave, retire with
// s_wait_tensorcnt, then publish with the workgroup barrier.
// =======================================================================
#define LDSROW 34  // 32 data dwords + 2 pad dwords (matches TDM pad fields)

__device__ __forceinline__ void tdm_load_tile_f32(
    unsigned lds_addr, const float* gaddr,
    unsigned tensor_d0, unsigned tensor_d1,
    unsigned tile_d1 /*rows*/, unsigned stride_elems)
{
    unsigned long long ga = (unsigned long long)(uintptr_t)gaddr;
    u32x4 g0;
    g0[0] = 1u;                                            // count=1, user mode
    g0[1] = lds_addr;                                      // lds_addr [63:32]
    g0[2] = (unsigned)(ga & 0xFFFFFFFFull);                // global_addr lo
    g0[3] = (unsigned)((ga >> 32) & 0x01FFFFFFull)         // global_addr hi
          | (2u << 30);                                    // type=2 ("image")
    u32x8 g1;
    g1[0] = (2u << 16)            // data_size = 4 bytes
          | (1u << 20)            // pad_enable
          | (4u << 22)            // pad_interval: 32 DWORDs
          | (1u << 25);           // pad_amount: 2 DWORDs
    g1[1] = (tensor_d0 & 0xFFFFu) << 16;                   // tensor_dim0[15:0]
    g1[2] = (tensor_d0 >> 16) | ((tensor_d1 & 0xFFFFu) << 16);
    g1[3] = (tensor_d1 >> 16) | (32u << 16);               // tile_dim0 = 32
    g1[4] = tile_d1 & 0xFFFFu;                             // tile_dim1 (tile_dim2=0)
    g1[5] = stride_elems;                                  // tensor_dim0_stride lo
    g1[6] = 0u;                                            // stride hi / dim1_stride lo
    g1[7] = 0u;
    asm volatile("tensor_load_to_lds %0, %1" :: "s"(g0), "s"(g1) : "memory");
}

// =======================================================================
// Generic GEMM:  C[M,N] = act( A[M,K] @ op(W) + bias ),
//   transB=1:  op(W)[k][n] = W[n*ldw + k]  (torch nn.Linear (N,K))
//              -> double-buffered TDM pipeline: DMA(t+1) overlaps WMMA(t);
//                 s_wait_tensorcnt(2) retires exactly the current tile pair
//                 (tensor ops complete in order per wave).
//   transB=0:  op(W)[k][n] = W[k*ldw + n]  (plain (K,N)) -> manual staging.
// Tile: BM=64 x BN=128 per 256-thread block (8 waves, 2x2 frags each).
// Both LDS tiles are K-contiguous per row so every fragment is a pair of
// wide ds loads + v_cvt_pk_bf16_f32.
// =======================================================================
#define BM 64
#define BN 128
#define BK 32

__global__ __launch_bounds__(256)
void gemm_bias_kernel(const float* __restrict__ A, int lda,
                      const float* __restrict__ W, int ldw, int transB,
                      const float* __restrict__ bias,
                      float* __restrict__ C, int ldc,
                      int M, int N, int K, int relu)
{
    __shared__ float sAf[2 * BM * LDSROW];   // ping-pong [m][k] k-contiguous
    __shared__ float sBf[2 * BN * LDSROW];   // ping-pong [n][k] k-contiguous

    const int tid  = threadIdx.x;
    const int lane = tid & 31;
    const int wave = tid >> 5;
    const int wr   = wave >> 2;   // 0..1 -> 32-row strip
    const int wc   = wave & 3;    // 0..3 -> 32-col strip
    const int m0   = blockIdx.y * BM;
    const int n0   = blockIdx.x * BN;
    const int l15  = lane & 15;
    const int halfA = (lane & 16) ? 16 : 0;   // A map: lanes 0-15 K 0..15
    const int halfB = (lane & 16) ? 16 : 0;   // B map: lanes 0-15 K 0..15

    const int ntiles = (K + BK - 1) / BK;

    v8f acc[2][2] = {};

    // ---- TDM prologue: wave0 kicks off tile 0 into buffer 0 ----
    if (transB && wave == 0) {
        tdm_load_tile_f32((unsigned)(uintptr_t)&sAf[0],
                          A + (size_t)m0 * lda,
                          (unsigned)K, (unsigned)(M - m0), BM, (unsigned)lda);
        tdm_load_tile_f32((unsigned)(uintptr_t)&sBf[0],
                          W + (size_t)n0 * ldw,
                          (unsigned)K, (unsigned)(N - n0), BN, (unsigned)ldw);
    }

    for (int t = 0; t < ntiles; ++t) {
        const int k0 = t * BK;
        float* bufA = &sAf[(size_t)(t & 1) * (BM * LDSROW)];
        float* bufB = &sBf[(size_t)(t & 1) * (BN * LDSROW)];

        // everyone finished computing tile t-1 -> buffer (t+1)&1 is free
        __syncthreads();

        if (transB) {
            if (wave == 0) {
                if (t + 1 < ntiles) {
                    const int k1 = k0 + BK;
                    float* nxtA = &sAf[(size_t)((t + 1) & 1) * (BM * LDSROW)];
                    float* nxtB = &sBf[(size_t)((t + 1) & 1) * (BN * LDSROW)];
                    tdm_load_tile_f32((unsigned)(uintptr_t)nxtA,
                                      A + (size_t)m0 * lda + k1,
                                      (unsigned)(K - k1), (unsigned)(M - m0),
                                      BM, (unsigned)lda);
                    tdm_load_tile_f32((unsigned)(uintptr_t)nxtB,
                                      W + (size_t)n0 * ldw + k1,
                                      (unsigned)(K - k1), (unsigned)(N - n0),
                                      BN, (unsigned)ldw);
                    // pair(t) done when only pair(t+1) (2 ops) remains in flight
                    __builtin_amdgcn_s_wait_tensorcnt(2);
                } else {
                    __builtin_amdgcn_s_wait_tensorcnt(0);
                }
            }
        } else {
            // ---- manual staging for (K,N) weights ----
            for (int i = tid; i < BM * BK; i += 256) {
                int r = i >> 5, c = i & 31;
                int gk = k0 + c;
                bufA[r * LDSROW + c] = (gk < K) ? A[(size_t)(m0 + r) * lda + gk] : 0.0f;
            }
            for (int i = tid; i < BN * BK; i += 256) {
                int n = i >> 5, c = i & 31;
                int gk = k0 + c;
                bufB[n * LDSROW + c] = (gk < K) ? W[(size_t)gk * ldw + (n0 + n)] : 0.0f;
            }
            if ((k0 + BK) < K)
                __builtin_prefetch(&A[(size_t)(m0 + (tid >> 2)) * lda + k0 + BK], 0, 1);
        }

        // publish tile t
        __syncthreads();

        // ---- build fragments (k = 16*(lane/16) + s for both A and B) + convert f32->bf16 ----
        v16bf afr[2], bfr[2];
#pragma unroll
        for (int tt = 0; tt < 2; ++tt) {
            const float* ar = &bufA[(size_t)(wr * 32 + tt * 16 + l15) * LDSROW];
            const float* br = &bufB[(size_t)(wc * 32 + tt * 16 + l15) * LDSROW];
#pragma unroll
            for (int i = 0; i < 8; ++i) {
                const int ka = halfA + 2 * i;
                afr[tt][2 * i]     = to_bf16(ar[ka]);
                afr[tt][2 * i + 1] = to_bf16(ar[ka + 1]);
                const int kbb = halfB + 2 * i;
                bfr[tt][2 * i]     = to_bf16(br[kbb]);
                bfr[tt][2 * i + 1] = to_bf16(br[kbb + 1]);
            }
        }
#pragma unroll
        for (int mt = 0; mt < 2; ++mt)
#pragma unroll
            for (int nt = 0; nt < 2; ++nt)
                acc[mt][nt] = wmma_bf16(afr[mt], bfr[nt], acc[mt][nt]);
    }

    // ---- epilogue: bias + optional ReLU, scatter per C/D layout ----
#pragma unroll
    for (int mt = 0; mt < 2; ++mt) {
#pragma unroll
        for (int nt = 0; nt < 2; ++nt) {
            const int n = n0 + wc * 32 + nt * 16 + l15;
            const float bv = bias ? bias[n] : 0.0f;
#pragma unroll
            for (int r = 0; r < 8; ++r) {
                const int m = m0 + wr * 32 + mt * 16 + r + ((lane & 16) ? 8 : 0);
                float v = acc[mt][nt][r] + bv;
                if (relu) v = fmaxf(v, 0.0f);
                C[(size_t)m * ldc + n] = v;
            }
        }
    }
}

// =======================================================================
// Fused attention: per-(b, h, 16-query block) workgroup.
//   S = softmax(scale * Q K^T + causal_mask);  O = S V
// Scores tile (16 x Sk) lives in LDS; both matmuls use WMMA bf16.
// =======================================================================
#define QT 16
#define SKMAX 512

__global__ __launch_bounds__(256)
void attn_kernel(const float* __restrict__ Qb, int ldq,
                 const float* __restrict__ Kb, int ldk,
                 const float* __restrict__ Vb, int ldv,
                 float* __restrict__ Ob, int ldo,
                 int Sq, int Sk, int causal)
{
    __shared__ float sS[QT][SKMAX + 4];
    __shared__ float sRed[QT][16];

    const int b    = blockIdx.z;
    const int h    = blockIdx.y;
    const int q0   = blockIdx.x * QT;
    const int tid  = threadIdx.x;
    const int lane = tid & 31;
    const int wave = tid >> 5;
    const int l15  = lane & 15;
    const int halfA = (lane & 16) ? 16 : 0;
    const int halfB = (lane & 16) ? 16 : 0;

    const float* Q  = Qb + (size_t)b * Sq * ldq + h * HDim;
    const float* Kp = Kb + (size_t)b * Sk * ldk + h * HDim;
    const float* Vp = Vb + (size_t)b * Sk * ldv + h * HDim;
    float*       O  = Ob + (size_t)b * Sq * ldo + h * HDim;

    const float scale = rsqrtf((float)HDim);

    // ---------- phase 1: scores = scale * Q K^T (+ causal mask) ----------
    for (int nt = wave; nt < (Sk >> 4); nt += 8) {
        v8f acc = {};
        const size_t qrow = (size_t)(q0 + l15) * ldq;
        const size_t krow = (size_t)(nt * 16 + l15) * ldk;
        for (int k0 = 0; k0 < HDim; k0 += 32) {
            v16bf af, bf;
#pragma unroll
            for (int i = 0; i < 8; ++i) {
                const int ka = halfA + 2 * i;
                af[2 * i]     = to_bf16(Q[qrow + k0 + ka]);
                af[2 * i + 1] = to_bf16(Q[qrow + k0 + ka + 1]);
                const int kb = halfB + 2 * i;
                bf[2 * i]     = to_bf16(Kp[krow + k0 + kb]);
                bf[2 * i + 1] = to_bf16(Kp[krow + k0 + kb + 1]);
            }
            acc = wmma_bf16(af, bf, acc);
        }
#pragma unroll
        for (int r = 0; r < 8; ++r) {
            const int m = r + ((lane & 16) ? 8 : 0);
            const int n = nt * 16 + l15;
            float v = acc[r] * scale;
            if (causal && (n > q0 + m)) v = NEG;
            sS[m][n] = v;
        }
    }
    __syncthreads();

    // ---------- phase 2: row softmax (16 threads per row) ----------
    {
        const int row  = tid >> 4;      // 0..15
        const int part = tid & 15;      // 0..15
        const int cpp  = Sk >> 4;       // cols per part (32)
        const int c0   = part * cpp;

        float mx = -3.4e38f;
        for (int c = 0; c < cpp; ++c) mx = fmaxf(mx, sS[row][c0 + c]);
        sRed[row][part] = mx;
        __syncthreads();
        float rmx = -3.4e38f;
        for (int i = 0; i < 16; ++i) rmx = fmaxf(rmx, sRed[row][i]);
        __syncthreads();

        float sum = 0.0f;
        for (int c = 0; c < cpp; ++c) {
            float e = __expf(sS[row][c0 + c] - rmx);
            sS[row][c0 + c] = e;
            sum += e;
        }
        sRed[row][part] = sum;
        __syncthreads();
        float rs = 0.0f;
        for (int i = 0; i < 16; ++i) rs += sRed[row][i];
        const float inv = 1.0f / rs;
        for (int c = 0; c < cpp; ++c) sS[row][c0 + c] *= inv;
    }
    __syncthreads();

    // ---------- phase 3: O = P V ----------
    for (int nt = wave; nt < (HDim >> 4); nt += 8) {
        v8f acc = {};
        const int n = nt * 16 + l15;
        for (int k0 = 0; k0 < Sk; k0 += 32) {
            v16bf af, bf;
#pragma unroll
            for (int i = 0; i < 8; ++i) {
                const int ka = halfA + 2 * i;
                af[2 * i]     = to_bf16(sS[l15][k0 + ka]);
                af[2 * i + 1] = to_bf16(sS[l15][k0 + ka + 1]);
                const int kb = halfB + 2 * i;
                bf[2 * i]     = to_bf16(Vp[(size_t)(k0 + kb)     * ldv + n]);
                bf[2 * i + 1] = to_bf16(Vp[(size_t)(k0 + kb + 1) * ldv + n]);
            }
            acc = wmma_bf16(af, bf, acc);
        }
#pragma unroll
        for (int r = 0; r < 8; ++r) {
            const int m = r + ((lane & 16) ? 8 : 0);
            O[(size_t)(q0 + m) * ldo + n] = acc[r];
        }
    }
}

// =======================================================================
// Fused residual-add + LayerNorm (one row per block), safe in-place.
// =======================================================================
__global__ __launch_bounds__(256)
void add_layernorm_kernel(const float* __restrict__ X, const float* __restrict__ R,
                          const float* __restrict__ g, const float* __restrict__ bta,
                          float* __restrict__ Y, int C)
{
    __shared__ float red[256];
    const int row = blockIdx.x;
    const float* xr = X + (size_t)row * C;
    const float* rr = R + (size_t)row * C;
    float*       yr = Y + (size_t)row * C;

    float vals[4];
    const int nper = (C + 255) >> 8;
    float s = 0.0f;
    for (int j = 0; j < nper; ++j) {
        int c = threadIdx.x + j * 256;
        float v = (c < C) ? (xr[c] + rr[c]) : 0.0f;
        vals[j] = v;
        s += v;
    }
    red[threadIdx.x] = s; __syncthreads();
    for (int st = 128; st > 0; st >>= 1) {
        if (threadIdx.x < st) red[threadIdx.x] += red[threadIdx.x + st];
        __syncthreads();
    }
    const float mu = red[0] / C;
    __syncthreads();

    float s2 = 0.0f;
    for (int j = 0; j < nper; ++j) {
        int c = threadIdx.x + j * 256;
        float d = vals[j] - mu;
        if (c < C) s2 += d * d;
    }
    red[threadIdx.x] = s2; __syncthreads();
    for (int st = 128; st > 0; st >>= 1) {
        if (threadIdx.x < st) red[threadIdx.x] += red[threadIdx.x + st];
        __syncthreads();
    }
    const float inv = rsqrtf(red[0] / C + 1e-5f);

    for (int j = 0; j < nper; ++j) {
        int c = threadIdx.x + j * 256;
        if (c < C) yr[c] = (vals[j] - mu) * inv * g[c] + bta[c];
    }
}

// =======================================================================
// Scheduled-sampling mix: out = where(sched_rand[s,b] >= steps/2e5, labels, dec1)
// =======================================================================
__global__ __launch_bounds__(256)
void mix_kernel(const float* __restrict__ labels, const float* __restrict__ dec1,
                const float* __restrict__ sched_rand, const int* __restrict__ steps,
                float* __restrict__ out)
{
    const size_t total = (size_t)Mrows * E;
    size_t i = (size_t)blockIdx.x * 256 + threadIdx.x;
    if (i >= total) return;
    const size_t bs = i / E;
    const int s = (int)(bs % Ss);
    const int b = (int)(bs / Ss);
    const float thr = (float)steps[0] / 200000.0f;
    const bool keep = sched_rand[(size_t)s * Bb + b] >= thr;
    out[i] = keep ? labels[i] : dec1[i];
}

// =======================================================================
// Host-side orchestration
// =======================================================================
static inline void launch_gemm(const float* A, int lda, const float* W, int ldw,
                               int transB, const float* bias, float* C, int ldc,
                               int M, int N, int K, int relu, hipStream_t st)
{
    dim3 g(N / BN, M / BM);
    gemm_bias_kernel<<<g, dim3(256), 0, st>>>(A, lda, W, ldw, transB, bias, C, ldc, M, N, K, relu);
}

struct DecParams {
    const float *sa_in_w, *sa_in_b, *sa_out_w, *sa_out_b;
    const float *ca_in_w, *ca_in_b, *ca_out_w, *ca_out_b;
    const float *w1, *b1, *w2, *b2;
    const float *g1, *be1, *g2, *be2, *g3, *be3;
    const float *mem;
    float *qkv, *t1, *t2, *ffn, *kvb;
};

static void run_decoder(float* x, const DecParams& p, hipStream_t st)
{
    const dim3 attn_grid(Ss / QT, Hh, Bb);
    for (int l = 0; l < Lnum; ++l) {
        const float* saw  = p.sa_in_w  + (size_t)l * 3 * E * E;
        const float* sab  = p.sa_in_b  + (size_t)l * 3 * E;
        const float* sow  = p.sa_out_w + (size_t)l * E * E;
        const float* sob  = p.sa_out_b + (size_t)l * E;
        const float* caw  = p.ca_in_w  + (size_t)l * 3 * E * E;
        const float* cab  = p.ca_in_b  + (size_t)l * 3 * E;
        const float* cow  = p.ca_out_w + (size_t)l * E * E;
        const float* cob  = p.ca_out_b + (size_t)l * E;

        // ---- self-attention ----
        launch_gemm(x, E, saw, E, 1, sab, p.qkv, 3 * E, Mrows, 3 * E, E, 0, st);
        attn_kernel<<<attn_grid, dim3(256), 0, st>>>(p.qkv, 3 * E,
                                                     p.qkv + E, 3 * E,
                                                     p.qkv + 2 * E, 3 * E,
                                                     p.t1, E, Ss, Ss, /*causal=*/1);
        launch_gemm(p.t1, E, sow, E, 1, sob, p.t2, E, Mrows, E, E, 0, st);
        add_layernorm_kernel<<<Mrows, 256, 0, st>>>(x, p.t2, p.g1 + l * E, p.be1 + l * E, x, E);

        // ---- cross-attention ----
        launch_gemm(x, E, caw, E, 1, cab, p.t1, E, Mrows, E, E, 0, st);                       // Q
        launch_gemm(p.mem, E, caw + (size_t)E * E, E, 1, cab + E, p.kvb, 2 * E,
                    Mrows, 2 * E, E, 0, st);                                                  // K,V
        attn_kernel<<<attn_grid, dim3(256), 0, st>>>(p.t1, E,
                                                     p.kvb, 2 * E,
                                                     p.kvb + E, 2 * E,
                                                     p.qkv, E, Ss, Ss, /*causal=*/0);
        launch_gemm(p.qkv, E, cow, E, 1, cob, p.t2, E, Mrows, E, E, 0, st);
        add_layernorm_kernel<<<Mrows, 256, 0, st>>>(x, p.t2, p.g2 + l * E, p.be2 + l * E, x, E);

        // ---- FFN ----
        launch_gemm(x, E, p.w1 + (size_t)l * FFdim * E, E, 1, p.b1 + l * FFdim,
                    p.ffn, FFdim, Mrows, FFdim, E, /*relu=*/1, st);
        launch_gemm(p.ffn, FFdim, p.w2 + (size_t)l * E * FFdim, FFdim, 1, p.b2 + l * E,
                    p.t2, E, Mrows, E, FFdim, 0, st);
        add_layernorm_kernel<<<Mrows, 256, 0, st>>>(x, p.t2, p.g3 + l * E, p.be3 + l * E, x, E);
    }
}

extern "C" void kernel_launch(void* const* d_in, const int* in_sizes, int n_in,
                              void* d_out, int out_size, void* d_ws, size_t ws_size,
                              hipStream_t stream)
{
    // inputs per setup_inputs() order
    const float* batch      = (const float*)d_in[0];
    const float* labels     = (const float*)d_in[1];
    const float* sched_rand = (const float*)d_in[2];
    const int*   steps      = (const int*)  d_in[3];
    const float* W_pre = (const float*)d_in[4];
    const float* b_pre = (const float*)d_in[5];
    const float* W_enc = (const float*)d_in[6];
    const float* b_enc = (const float*)d_in[7];
    const float* W_e2d = (const float*)d_in[8];
    const float* b_e2d = (const float*)d_in[9];

    DecParams p;
    p.sa_in_w  = (const float*)d_in[10]; p.sa_in_b  = (const float*)d_in[11];
    p.sa_out_w = (const float*)d_in[12]; p.sa_out_b = (const float*)d_in[13];
    p.ca_in_w  = (const float*)d_in[14]; p.ca_in_b  = (const float*)d_in[15];
    p.ca_out_w = (const float*)d_in[16]; p.ca_out_b = (const float*)d_in[17];
    p.w1  = (const float*)d_in[18]; p.b1  = (const float*)d_in[19];
    p.w2  = (const float*)d_in[20]; p.b2  = (const float*)d_in[21];
    p.g1  = (const float*)d_in[22]; p.be1 = (const float*)d_in[23];
    p.g2  = (const float*)d_in[24]; p.be2 = (const float*)d_in[25];
    p.g3  = (const float*)d_in[26]; p.be3 = (const float*)d_in[27];

    // workspace carving (floats)
    const size_t NBS = (size_t)Mrows;
    float* ws   = (float*)d_ws;
    float* mem  = ws;                     // NBS*E
    float* qkv  = mem  + NBS * E;         // NBS*3E
    float* t1   = qkv  + NBS * 3 * E;     // NBS*E
    float* t2   = t1   + NBS * E;         // NBS*E
    float* ffn  = t2   + NBS * E;         // NBS*FF
    float* dec1 = ffn  + NBS * FFdim;     // NBS*E
    float* enc  = dec1 + NBS * E;         // NBS*ENC
    float* kvb  = enc  + NBS * ENC;       // NBS*2E
    p.mem = mem; p.qkv = qkv; p.t1 = t1; p.t2 = t2; p.ffn = ffn; p.kvb = kvb;

    // 1) prenet + (simplified) encoder + enc_2_dec  -> mem
    launch_gemm(batch, INDIM, W_pre, E, /*transB=*/0, b_pre, t2, E, Mrows, E, INDIM, 0, stream);
    launch_gemm(t2, E, W_enc, ENC, 0, b_enc, enc, ENC, Mrows, ENC, E, 0, stream);
    launch_gemm(enc, ENC, W_e2d, E, 0, b_e2d, mem, E, Mrows, E, ENC, 0, stream);

    // 2) first decoder pass: dec1 = decoder(labels, mem)
    hipMemcpyAsync(dec1, labels, NBS * E * sizeof(float), hipMemcpyDeviceToDevice, stream);
    run_decoder(dec1, p, stream);

    // 3) scheduled-sampling mix -> d_out, then second decoder pass in place
    float* out = (float*)d_out;
    {
        const size_t total = NBS * (size_t)E;
        const int blocks = (int)((total + 255) / 256);
        mix_kernel<<<blocks, dim3(256), 0, stream>>>(labels, dec1, sched_rand, steps, out);
    }
    run_decoder(out, p, stream);
}
